// HoloConvNeXtBlock_v3_74320114090558
// MI455X (gfx1250) — matrix operand order repaired and checked by reading, not compiled, not run on hardware
//
#include <hip/hip_runtime.h>
#include <math.h>

// ---------------- types ----------------
typedef __bf16  bf16x8  __attribute__((ext_vector_type(8)));
typedef __bf16  bf16x16 __attribute__((ext_vector_type(16)));
typedef float   f32x8   __attribute__((ext_vector_type(8)));
typedef unsigned short u16;

// Problem constants
#define C0     96
#define C4     384
#define HH     40
#define WW     40
#define P      1600          // spatial positions
#define S      100           // phi slices
#define NSLOT  102           // s, bt, 100 phi
#define MTOT   (NSLOT*P)     // 163200 rows, multiple of 16
#define MT_ALL (MTOT/16)     // 10200 M-tiles
#define NT1    (C4/16)       // 24 N-tiles for GEMM1
#define NT2    (C0/16)       // 6  N-tiles for GEMM2
#define KT1    (C0/32)       // 3  K-steps GEMM1
#define KT2    (C4/32)       // 12 K-steps GEMM2
#define MW1    4             // M-tiles per wave, GEMM1 bt/phi
#define MW2    2             // M-tiles per wave, GEMM2

static __device__ __forceinline__ u16 f2bf(float x) {
  union { float f; unsigned u; } v; v.f = x;
  unsigned r = v.u + 0x7FFFu + ((v.u >> 16) & 1u);   // RNE
  return (u16)(r >> 16);
}
static __device__ __forceinline__ bf16x16 cat16(bf16x8 lo, bf16x8 hi) {
  return __builtin_shufflevector(lo, hi, 0,1,2,3,4,5,6,7,8,9,10,11,12,13,14,15);
}
// A tile per ISA layout: lane m = l&15, K in {kb..kb+7, kb+16..kb+23}, kb = 8*(l>=16)
static __device__ __forceinline__ bf16x16 load_a_tile(const u16* base, int row_stride,
                                                      int r0, int kofs, int lane) {
  int m  = lane & 15;
  int kb = (lane >> 4) * 8;
  const u16* p = base + (size_t)(r0 + m) * row_stride + kofs + kb;
  bf16x8 lo = *(const bf16x8*)(p);
  bf16x8 hi = *(const bf16x8*)(p + 16);
  return cat16(lo, hi);
}
// B tile from packed layout (global)
static __device__ __forceinline__ bf16x16 load_b_tile(const u16* pack, int tile_id, int lane) {
  const u16* p = pack + ((size_t)tile_id * 32 + lane) * 16;
  bf16x8 lo = *(const bf16x8*)(p);
  bf16x8 hi = *(const bf16x8*)(p + 8);
  return cat16(lo, hi);
}
// B tile from packed layout staged in LDS
static __device__ __forceinline__ bf16x16 load_b_lds(const u16* sm, int tile_id, int lane) {
  const u16* p = sm + tile_id * 512 + lane * 16;
  bf16x8 lo = *(const bf16x8*)(p);
  bf16x8 hi = *(const bf16x8*)(p + 8);
  return cat16(lo, hi);
}
static __device__ __forceinline__ f32x8 wmma_bf16(bf16x16 a, bf16x16 b, f32x8 c) {
  return __builtin_amdgcn_wmma_f32_16x16x32_bf16(false, a, false, b, (short)0, c, false, false);
}

// ---------------- kernel: pack W1/W2 into WMMA B layout ----------------
__global__ void pack_weights(const float* __restrict__ w1, const float* __restrict__ w2,
                             u16* __restrict__ bp1, u16* __restrict__ bp2) {
  int t = blockIdx.x * blockDim.x + threadIdx.x;
  if (t < KT1 * NT1 * 32 * 16) {                 // 36864
    int i = t & 15, lane = (t >> 4) & 31, rem = t >> 9;
    int nt = rem % NT1, kt = rem / NT1;
    int n = lane & 15, kb = (lane >> 4) * 8;
    int k = kt * 32 + kb + i + (i >= 8 ? 8 : 0); // c index 0..95
    int o = nt * 16 + n;                         // 0..383
    bp1[t] = f2bf(w1[o * C0 + k]);
  } else if (t < 2 * KT1 * NT1 * 32 * 16) {
    int t2 = t - KT1 * NT1 * 32 * 16;
    int i = t2 & 15, lane = (t2 >> 4) & 31, rem = t2 >> 9;
    int nt = rem % NT2, kt = rem / NT2;          // kt 0..11
    int n = lane & 15, kb = (lane >> 4) * 8;
    int k = kt * 32 + kb + i + (i >= 8 ? 8 : 0); // o index 0..383
    int c = nt * 16 + n;                         // 0..95
    bp2[t2] = f2bf(w2[c * C4 + k]);
  }
}

// ---------------- kernel: depthwise 7x7 on signal & beta ----------------
__global__ void dwconv_sb(const float* __restrict__ sig, const float* __restrict__ beta,
                          const float* __restrict__ dww, const float* __restrict__ dwb,
                          float* __restrict__ sconv, float* __restrict__ btconv) {
  int t = blockIdx.x * blockDim.x + threadIdx.x;
  if (t >= C0 * P) return;
  int pos = t % P, c = t / P;
  int h = pos / WW, w = pos % WW;
  const float* wp = dww + c * 49;
  const float* sp = sig  + (size_t)c * P;
  const float* bp = beta + (size_t)c * P;
  float as = dwb[c], ab = dwb[c];
  #pragma unroll
  for (int kh = 0; kh < 7; ++kh) {
    int ih = h + kh - 3; if ((unsigned)ih >= (unsigned)HH) continue;
    #pragma unroll
    for (int kw = 0; kw < 7; ++kw) {
      int iw = w + kw - 3; if ((unsigned)iw >= (unsigned)WW) continue;
      float wt = wp[kh * 7 + kw];
      as += sp[ih * WW + iw] * wt;
      ab += bp[ih * WW + iw] * wt;
    }
  }
  sconv[pos * C0 + c]  = as;
  btconv[pos * C0 + c] = ab;
}

// ---------------- kernel: LayerNorm per position; emit A rows for s & bt ----------------
__global__ void ln_pack(const float* __restrict__ sconv, const float* __restrict__ btconv,
                        const float* __restrict__ lnw, const float* __restrict__ lnb,
                        u16* __restrict__ A, float* __restrict__ scaleArr) {
  int pos = blockIdx.x;
  int tid = threadIdx.x;                       // 128 threads
  __shared__ float sh[128], sh2[128];
  float s = 0.f, bt = 0.f;
  if (tid < C0) { s = sconv[pos * C0 + tid]; bt = btconv[pos * C0 + tid]; }
  sh[tid]  = (tid < C0) ? s     : 0.f;
  sh2[tid] = (tid < C0) ? s * s : 0.f;
  __syncthreads();
  for (int off = 64; off > 0; off >>= 1) {
    if (tid < off) { sh[tid] += sh[tid + off]; sh2[tid] += sh2[tid + off]; }
    __syncthreads();
  }
  float u   = sh[0]  * (1.0f / C0);
  float var = sh2[0] * (1.0f / C0) - u * u;
  float inv = rsqrtf(var + 1e-6f);
  if (tid < C0) {
    float sc  = lnw[tid] * inv;
    float sht = lnb[tid] - u * sc;
    A[(size_t)pos * C0 + tid]       = f2bf(s  * sc + sht);   // slot 0 (s)
    A[(size_t)(P + pos) * C0 + tid] = f2bf(bt * sc + sht);   // slot 1 (bt)
    scaleArr[pos * C0 + tid] = sc;
  }
}

// ---------------- kernel: depthwise 7x7 on phi, LN-scale, emit A rows ----------------
__global__ void dwconv_phi(const float* __restrict__ phi, const float* __restrict__ dww,
                           const float* __restrict__ scaleArr, u16* __restrict__ A) {
  size_t t = (size_t)blockIdx.x * blockDim.x + threadIdx.x;
  if (t >= (size_t)C0 * P * S) return;
  int s = (int)(t % S);
  size_t r = t / S;
  int pos = (int)(r % P);
  int c   = (int)(r / P);
  int h = pos / WW, w = pos % WW;
  const float* wp = dww + c * 49;
  const float* pp = phi + (size_t)c * P * S;
  float acc = 0.f;
  #pragma unroll
  for (int kh = 0; kh < 7; ++kh) {
    int ih = h + kh - 3; if ((unsigned)ih >= (unsigned)HH) continue;
    #pragma unroll
    for (int kw = 0; kw < 7; ++kw) {
      int iw = w + kw - 3; if ((unsigned)iw >= (unsigned)WW) continue;
      acc += pp[(size_t)(ih * WW + iw) * S + s] * wp[kh * 7 + kw];
    }
  }
  acc *= scaleArr[pos * C0 + c];
  A[((size_t)(2 + s) * P + pos) * C0 + c] = f2bf(acc);       // slot 2+s
}

// ---------------- kernel: GEMM1 over s-rows (+b1, GELU, ratio) ----------------
__global__ void gemm1_s(const u16* __restrict__ A, const u16* __restrict__ bp1,
                        const float* __restrict__ b1,
                        u16* __restrict__ Hh, float* __restrict__ ratio) {
  int wv = (blockIdx.x * blockDim.x + threadIdx.x) >> 5;
  int lane = threadIdx.x & 31;
  int mt = wv / 6, ng = wv % 6;                 // 4 N-tiles per wave
  if (mt >= P / 16) return;
  int r0 = mt * 16;
  f32x8 acc[4] = {};
  #pragma unroll
  for (int kt = 0; kt < KT1; ++kt) {
    bf16x16 a = load_a_tile(A, C0, r0, kt * 32, lane);
    #pragma unroll
    for (int j = 0; j < 4; ++j) {
      bf16x16 b = load_b_tile(bp1, kt * NT1 + (ng * 4 + j), lane);
      acc[j] = wmma_bf16(a, b, acc[j]);
    }
  }
  int n = lane & 15, mb = (lane >> 4) * 8;
  #pragma unroll
  for (int j = 0; j < 4; ++j) {
    int o = (ng * 4 + j) * 16 + n;
    float bias = b1[o];
    #pragma unroll
    for (int i = 0; i < 8; ++i) {
      int r = r0 + mb + i;                      // == pos
      float x = acc[j][i] + bias;
      float g = 0.5f * x * (1.0f + erff(x * 0.70710678118654752f));
      float rt = (fabsf(x) > 1e-6f) ? (g / x) : 0.0f;
      ratio[(size_t)r * C4 + o] = rt;
      Hh[(size_t)r * C4 + o]    = f2bf(g);
    }
  }
}

// ---- kernel: GEMM1 over bt+phi rows; B staged via LDS, register-resident ----
// Block = 8 waves, one N-group (4 N-tiles); each wave sweeps MW1 M-tiles.
__global__ void gemm1_r(const u16* __restrict__ A, const u16* __restrict__ bp1,
                        const float* __restrict__ b1, const float* __restrict__ ratio,
                        u16* __restrict__ Hh) {
  __shared__ __align__(16) u16 smem[12 * 512];              // 12 KB: 3 kt x 4 nt tiles
  int tid = threadIdx.x;
  int ng     = blockIdx.x % 6;
  int blockm = blockIdx.x / 6;
  // cooperative stage of this N-group's 12 B tiles (3 chunks of 4 contiguous tiles)
  for (int i = tid; i < 768; i += 256) {                    // 768 x 16B = 12 KB
    int kt = i >> 8, off = i & 255;
    *(uint4*)(smem + kt * 2048 + off * 8) =
      *(const uint4*)(bp1 + ((size_t)(kt * NT1 + ng * 4) * 512) + off * 8);
  }
  __syncthreads();
  int lane = tid & 31, waveid = tid >> 5;
  // hoist all 12 B tiles into registers (LDS -> ds_load_b128 x24, once per wave)
  bf16x16 Breg[KT1][4];
  #pragma unroll
  for (int kt = 0; kt < KT1; ++kt)
    #pragma unroll
    for (int j = 0; j < 4; ++j)
      Breg[kt][j] = load_b_lds(smem, kt * 4 + j, lane);

  int n = lane & 15, mb = (lane >> 4) * 8;
  for (int mi = 0; mi < MW1; ++mi) {
    int mt = (P / 16) + blockm * (8 * MW1) + waveid * MW1 + mi;   // wave-uniform
    if (mt >= MT_ALL) continue;
    int r0 = mt * 16;
    bool has_bias = (r0 < 2 * P);               // bt region only
    f32x8 acc[4] = {};
    #pragma unroll
    for (int kt = 0; kt < KT1; ++kt) {
      bf16x16 a = load_a_tile(A, C0, r0, kt * 32, lane);    // 2 b128 loads
      #pragma unroll
      for (int j = 0; j < 4; ++j)
        acc[j] = wmma_bf16(a, Breg[kt][j], acc[j]);
    }
    #pragma unroll
    for (int j = 0; j < 4; ++j) {
      int o = (ng * 4 + j) * 16 + n;
      float bias = has_bias ? b1[o] : 0.0f;
      #pragma unroll
      for (int i = 0; i < 8; ++i) {
        int r = r0 + mb + i;
        int pos = r % P;
        float x = acc[j][i] + bias;
        Hh[(size_t)r * C4 + o] = f2bf(x * ratio[(size_t)pos * C4 + o]);
      }
    }
  }
}

// ---- kernel: GEMM2; W2 pack staged in LDS in two 36KB phases ----
// Each wave: MW2 M-tiles x all 6 N-tiles, acc held across phases.
__global__ void gemm2(const u16* __restrict__ Hh, const u16* __restrict__ bp2,
                      const float* __restrict__ b2, const float* __restrict__ gamma,
                      const float* __restrict__ sig, const float* __restrict__ beta,
                      const float* __restrict__ phi,
                      float* __restrict__ outS, float* __restrict__ outPH,
                      float* __restrict__ outBT) {
  __shared__ __align__(16) u16 smem[36 * 512];              // 36 KB: 6 kt x 6 nt tiles
  int tid = threadIdx.x, lane = tid & 31, waveid = tid >> 5;
  int wv = blockIdx.x * 8 + waveid;
  f32x8 acc[MW2][NT2] = {};
  for (int ph = 0; ph < 2; ++ph) {
    __syncthreads();                                        // protect LDS reuse
    for (int i = tid; i < 2304; i += 256)                   // 2304 x 16B = 36 KB
      ((uint4*)smem)[i] = ((const uint4*)(bp2 + (size_t)ph * 36 * 512))[i];
    __syncthreads();
    for (int mi = 0; mi < MW2; ++mi) {
      int mt = wv * MW2 + mi;                               // wave-uniform
      if (mt >= MT_ALL) continue;
      int r0 = mt * 16;
      #pragma unroll
      for (int k6 = 0; k6 < 6; ++k6) {
        int kt = ph * 6 + k6;
        bf16x16 a = load_a_tile(Hh, C4, r0, kt * 32, lane); // 2 b128 global
        #pragma unroll
        for (int j = 0; j < NT2; ++j) {
          bf16x16 b = load_b_lds(smem, k6 * NT2 + j, lane); // 2 b128 LDS
          acc[mi][j] = wmma_bf16(a, b, acc[mi][j]);
        }
      }
    }
  }
  int n = lane & 15, mb = (lane >> 4) * 8;
  for (int mi = 0; mi < MW2; ++mi) {
    int mt = wv * MW2 + mi;
    if (mt >= MT_ALL) continue;
    int r0 = mt * 16;
    bool has_bias = (r0 < 2 * P);               // s and bt regions
    int region = (r0 < P) ? 0 : (r0 < 2 * P ? 1 : 2);
    #pragma unroll
    for (int j = 0; j < NT2; ++j) {
      int c = j * 16 + n;
      float bias = has_bias ? b2[c] : 0.0f;
      float gm = gamma[c];
      #pragma unroll
      for (int i = 0; i < 8; ++i) {
        int r = r0 + mb + i;
        float y = acc[mi][j][i] + bias;
        if (region == 0) {
          outS[(size_t)c * P + r] = y * gm + sig[(size_t)c * P + r];
        } else if (region == 1) {
          int pos = r - P;
          outBT[(size_t)c * P + pos] = y * gm + beta[(size_t)c * P + pos];
        } else {
          int sidx = r / P - 2;
          int pos  = r % P;
          size_t idx = ((size_t)c * P + pos) * S + sidx;
          outPH[idx] = y * gm + phi[idx];
        }
      }
    }
  }
}

// ---------------- kernel: top-3 |phi_raw| indices (gamma-invariant ordering) ----------------
__global__ void topk3(const float* __restrict__ outPH, const float* __restrict__ phi,
                      float* __restrict__ pi) {
  int t = blockIdx.x * blockDim.x + threadIdx.x;
  if (t >= C0 * P) return;                      // t = c*P + pos
  const float* a = outPH + (size_t)t * S;
  const float* b = phi   + (size_t)t * S;
  float v0 = -1.f, v1 = -1.f, v2 = -1.f;
  int i0 = 0, i1 = 0, i2 = 0;
  for (int s = 0; s < S; ++s) {
    float d = fabsf(a[s] - b[s]);               // = |ph_raw| * gamma[c], same order
    if (d > v0)      { v2=v1; i2=i1; v1=v0; i1=i0; v0=d; i0=s; }
    else if (d > v1) { v2=v1; i2=i1; v1=d; i1=s; }
    else if (d > v2) { v2=d; i2=s; }
  }
  pi[(size_t)t * 3 + 0] = (float)i0;
  pi[(size_t)t * 3 + 1] = (float)i1;
  pi[(size_t)t * 3 + 2] = (float)i2;
}

// ---------------- host launch ----------------
extern "C" void kernel_launch(void* const* d_in, const int* in_sizes, int n_in,
                              void* d_out, int out_size, void* d_ws, size_t ws_size,
                              hipStream_t stream) {
  (void)in_sizes; (void)n_in; (void)out_size; (void)ws_size;
  const float* sig   = (const float*)d_in[0];
  const float* phi   = (const float*)d_in[1];
  const float* beta  = (const float*)d_in[2];
  const float* dww   = (const float*)d_in[3];
  const float* dwb   = (const float*)d_in[4];
  const float* lnw   = (const float*)d_in[5];
  const float* lnb   = (const float*)d_in[6];
  const float* w1    = (const float*)d_in[7];
  const float* b1    = (const float*)d_in[8];
  const float* w2    = (const float*)d_in[9];
  const float* b2    = (const float*)d_in[10];
  const float* gamma = (const float*)d_in[11];

  float* outS  = (float*)d_out;                             // [96,1600]
  float* outPH = outS + (size_t)C0 * P;                     // [96,1600,100]
  float* outBT = outPH + (size_t)C0 * P * S;                // [96,1600]
  float* outPI = outBT + (size_t)C0 * P;                    // [96,1600,3]

  char* ws = (char*)d_ws;
  size_t off = 0;
  auto alloc = [&](size_t bytes) { char* p = ws + off; off = (off + bytes + 255) & ~(size_t)255; return p; };
  u16*   A      = (u16*)  alloc((size_t)MTOT * C0 * 2);     // 31.3 MB
  u16*   Hh     = (u16*)  alloc((size_t)MTOT * C4 * 2);     // 125.3 MB
  float* ratio  = (float*)alloc((size_t)P * C4 * 4);        // 2.46 MB
  float* sconv  = (float*)alloc((size_t)P * C0 * 4);
  float* btconv = (float*)alloc((size_t)P * C0 * 4);
  float* scArr  = (float*)alloc((size_t)P * C0 * 4);
  u16*   bp1    = (u16*)  alloc((size_t)KT1 * NT1 * 32 * 16 * 2);
  u16*   bp2    = (u16*)  alloc((size_t)KT2 * NT2 * 32 * 16 * 2);

  // 1. weight packing (73728 elements)
  pack_weights<<<288, 256, 0, stream>>>(w1, w2, bp1, bp2);
  // 2. depthwise conv on signal & beta (153600 threads)
  dwconv_sb<<<600, 256, 0, stream>>>(sig, beta, dww, dwb, sconv, btconv);
  // 3. LayerNorm + pack s/bt rows (one block per position)
  ln_pack<<<P, 128, 0, stream>>>(sconv, btconv, lnw, lnb, A, scArr);
  // 4. depthwise conv on phi + LN scale (15.36M threads)
  dwconv_phi<<<60000, 256, 0, stream>>>(phi, dww, scArr, A);
  // 5. GEMM1 s-rows: 100 mtiles * 6 ngroups = 600 waves
  gemm1_s<<<75, 256, 0, stream>>>(A, bp1, b1, Hh, ratio);
  // 6. GEMM1 bt+phi rows: 6 ngroups x ceil(10100/32) m-blocks = 1896 blocks
  gemm1_r<<<6 * ((MTOT/16 - P/16 + 8*MW1 - 1) / (8*MW1)), 256, 0, stream>>>(A, bp1, b1, ratio, Hh);
  // 7. GEMM2: ceil(10200/16) = 638 blocks
  gemm2<<<(MT_ALL + 8*MW2 - 1) / (8*MW2), 256, 0, stream>>>(Hh, bp2, b2, gamma, sig, beta, phi,
                                                            outS, outPH, outBT);
  // 8. top-3 indices (153600 threads)
  topk3<<<600, 256, 0, stream>>>(outPH, phi, outPI);
}
